// KipfNet3_30210799960813
// MI455X (gfx1250) — matrix operand order repaired and checked by reading, not compiled, not run on hardware
//
#include <hip/hip_runtime.h>

typedef __attribute__((ext_vector_type(16))) _Float16 v16h;
typedef __attribute__((ext_vector_type(8)))  float    v8f;

#define N_NODES 100000
#define N_EDGES 1600000
#define BN_EPS 1e-5f

static inline long cdiv(long a, long b) { return (a + b - 1) / b; }

// ---------------------------------------------------------------------------
// Utility kernels
// ---------------------------------------------------------------------------
__global__ __launch_bounds__(256) void kzero(float* __restrict__ p, long n) {
    long i = (long)blockIdx.x * blockDim.x + threadIdx.x;
    if (i < n) p[i] = 0.0f;
}

// out[i] = -in[i]  (in == out allowed: elementwise)
__global__ __launch_bounds__(256) void knegcopy(float* __restrict__ out,
                                                const float* __restrict__ in, long n) {
    long i = (long)blockIdx.x * blockDim.x + threadIdx.x;
    if (i < n) out[i] = -in[i];
}

// out (N x ldc): col < d -> bias[col], padding cols -> 0
__global__ __launch_bounds__(256) void kinit_bias(float* __restrict__ out,
                                                  const float* __restrict__ bias,
                                                  int d, int ldc, long n) {
    long i = (long)blockIdx.x * blockDim.x + threadIdx.x;
    if (i >= n) return;
    int c = (int)(i % ldc);
    out[i] = (c < d) ? bias[c] : 0.0f;
}

__global__ __launch_bounds__(256) void kdeg(const int* __restrict__ dst,
                                            float* __restrict__ deg, int ne) {
    int e = blockIdx.x * blockDim.x + threadIdx.x;
    if (e < ne) atomicAdd(&deg[dst[e]], 1.0f);
}

__global__ __launch_bounds__(256) void kdinv(const float* __restrict__ deg,
                                             float* __restrict__ dinv, int n) {
    int i = blockIdx.x * blockDim.x + threadIdx.x;
    if (i < n) {
        float d = deg[i];
        dinv[i] = (d > 0.0f) ? rsqrtf(fmaxf(d, 1.0f)) : 0.0f;
    }
}

__global__ __launch_bounds__(256) void kwnorm(const int* __restrict__ src,
                                              const int* __restrict__ dst,
                                              const float* __restrict__ dinv,
                                              float* __restrict__ wn, int ne) {
    int e = blockIdx.x * blockDim.x + threadIdx.x;
    if (e < ne) wn[e] = -dinv[src[e]] * dinv[dst[e]];
}

// One thread per (edge, logical feature); rows have physical stride `ld`.
__global__ __launch_bounds__(256) void kscatter(const int* __restrict__ src,
                                                const int* __restrict__ dst,
                                                const float* __restrict__ wn,
                                                const float* __restrict__ T,
                                                float* __restrict__ out,
                                                int dinL, int ld, float coef, long total) {
    long i = (long)blockIdx.x * blockDim.x + threadIdx.x;
    if (i >= total) return;
    int e = (int)(i / dinL);
    int f = (int)(i - (long)e * dinL);
    float v = coef * wn[e] * T[(size_t)src[e] * ld + f];
    atomicAdd(&out[(size_t)dst[e] * ld + f], v);
}

// BN (eval) + ReLU over (N x ldc) buffer; only logical cols < d touched.
__global__ __launch_bounds__(256) void kbnrelu(float* __restrict__ h,
                                               const float* __restrict__ g,
                                               const float* __restrict__ b,
                                               const float* __restrict__ m,
                                               const float* __restrict__ v,
                                               int d, int ldc, long total) {
    long i = (long)blockIdx.x * blockDim.x + threadIdx.x;
    if (i >= total) return;
    int c = (int)(i % ldc);
    if (c >= d) return;
    float val = (h[i] - m[c]) * (g[c] * rsqrtf(v[c] + BN_EPS)) + b[c];
    h[i] = fmaxf(val, 0.0f);
}

// ---------------------------------------------------------------------------
// Weight pre-pack into a WMMA B-fragment layout in contiguous k order (same as A, not the ISA table order):
// per (hop k, k-step ks, n-tile tn, lane): 16 f16 = 16 consecutive K values
// of column ncol, K half selected by lane>>4.
// Index: (((k*KSTEPS + ks)*NTILE + tn)*32 + lane)*16 halfs.
// ---------------------------------------------------------------------------
template <int DINP, int DINL, int DOUT>
__global__ __launch_bounds__(128)
void kpackB(const float* __restrict__ W, _Float16* __restrict__ out, int K) {
    constexpr int NTILE = (DOUT + 15) / 16;
    constexpr int KSTEPS = DINP / 32;
    int idx = blockIdx.x * blockDim.x + threadIdx.x;
    int total = K * KSTEPS * NTILE * 32;
    if (idx >= total) return;
    int lane = idx & 31;
    int rest = idx >> 5;
    int tn = rest % NTILE; rest /= NTILE;
    int ks = rest % KSTEPS; rest /= KSTEPS;
    int k = rest;
    int hi = lane >> 4, lo = lane & 15;
    int n = tn * 16 + lo;
    v16h frag;
#pragma unroll
    for (int t = 0; t < 16; ++t) {
        int krow = ks * 32 + hi * 16 + t;
        float w = (n < DOUT && krow < DINL)
                      ? W[((size_t)k * DINL + krow) * DOUT + n] : 0.0f;
        frag[t] = (_Float16)w;
    }
    *(v16h*)(out + (size_t)idx * 16) = frag;
}

// ---------------------------------------------------------------------------
// WMMA GEMM-accumulate: C(N_NODES x LDC) += A(N_NODES x DIN) * Bpacked
// One wave32 per 16x16 output tile. All hot-loop loads unconditional.
// ---------------------------------------------------------------------------
template <int DIN, int DOUT, int LDC>
__global__ __launch_bounds__(128)
void kgemm_wmma(const float* __restrict__ A, const _Float16* __restrict__ Bp,
                float* __restrict__ C, int total_tiles) {
    constexpr int NTILE = (DOUT + 15) / 16;
    constexpr int KSTEPS = DIN / 32;
    constexpr bool GUARD = (LDC < NTILE * 16);   // only final layer (ldc=10)

    const int wave = threadIdx.x >> 5;
    const int lane = threadIdx.x & 31;
    const int tile = blockIdx.x * 4 + wave;      // wave-uniform
    if (tile >= total_tiles) return;             // uniform: EXEC all-1s below

    const int tm = tile / NTILE;
    const int tn = tile - tm * NTILE;
    const int hi = lane >> 4;
    const int lo = lane & 15;
    const int ncol = tn * 16 + lo;
    const float* __restrict__ arow = A + (size_t)(tm * 16 + lo) * DIN;

    v8f acc;
#pragma unroll
    for (int r = 0; r < 8; ++r) {
        int mrow = tm * 16 + r + 8 * hi;
        acc[r] = (!GUARD || ncol < DOUT) ? C[(size_t)mrow * LDC + ncol] : 0.0f;
    }

#pragma unroll
    for (int ks = 0; ks < KSTEPS; ++ks) {
        // A fragment: lane half hi needs floats [ks*32+hi*16, +16)
        const float4* pa0 = (const float4*)(arow + ks * 32 + hi * 16);
        const float4* pa1 = (const float4*)(arow + ks * 32 + hi * 16 + 8);
        float4 a0 = pa0[0], a1 = pa0[1];
        float4 a2 = pa1[0], a3 = pa1[1];
        v16h af;
        af[0]  = (_Float16)a0.x; af[1]  = (_Float16)a0.y;
        af[2]  = (_Float16)a0.z; af[3]  = (_Float16)a0.w;
        af[4]  = (_Float16)a1.x; af[5]  = (_Float16)a1.y;
        af[6]  = (_Float16)a1.z; af[7]  = (_Float16)a1.w;
        af[8]  = (_Float16)a2.x; af[9]  = (_Float16)a2.y;
        af[10] = (_Float16)a2.z; af[11] = (_Float16)a2.w;
        af[12] = (_Float16)a3.x; af[13] = (_Float16)a3.y;
        af[14] = (_Float16)a3.z; af[15] = (_Float16)a3.w;

        // B fragment: one unconditional 32-byte per-lane load from packed buffer
        v16h bf = *(const v16h*)(Bp + ((size_t)(ks * NTILE + tn) * 32 + lane) * 16);

        acc = __builtin_amdgcn_wmma_f32_16x16x32_f16(
            /*neg_a=*/false, af, /*neg_b=*/false, bf,
            /*c_mod=*/(short)0, acc, /*reuse_a=*/false, /*reuse_b=*/false);
    }

#pragma unroll
    for (int r = 0; r < 8; ++r) {
        int mrow = tm * 16 + r + 8 * hi;
        if (!GUARD || ncol < DOUT) C[(size_t)mrow * LDC + ncol] = acc[r];
    }
}

// ---------------------------------------------------------------------------
// Host-side layer driver
// ---------------------------------------------------------------------------
template <int DINP, int DINL, int DOUT, int LDC>
static void cheb_layer(const float* h, const float* W, const float* bias, int K,
                       float* out, const int* src, const int* dst, const float* wn,
                       float* bufA, float* bufB, _Float16* packB, hipStream_t stream) {
    constexpr int NTILE = (DOUT + 15) / 16;
    constexpr int KSTEPS = DINP / 32;
    const long nd = (long)N_NODES * DINP;
    const long nout = (long)N_NODES * LDC;
    const long escan = (long)N_EDGES * DINL;
    const size_t hopstride = (size_t)KSTEPS * NTILE * 32 * 16;

    // pre-pack all hop weights into WMMA B layout
    {
        int total = K * KSTEPS * NTILE * 32;
        kpackB<DINP, DINL, DOUT><<<(unsigned)cdiv(total, 128), 128, 0, stream>>>(W, packB, K);
    }
    const int total_tiles = (N_NODES / 16) * NTILE;  // 100000 = 6250*16 exactly
    dim3 gg((unsigned)cdiv(total_tiles, 4));

    // out = bias (padding cols = 0) ; out += T0 @ W0   (T0 = h)
    kinit_bias<<<(unsigned)cdiv(nout, 256), 256, 0, stream>>>(out, bias, DOUT, LDC, nout);
    kgemm_wmma<DINP, DOUT, LDC><<<gg, 128, 0, stream>>>(h, packB, out, total_tiles);

    // T1 = L_hat @ T0
    kzero<<<(unsigned)cdiv(nd, 256), 256, 0, stream>>>(bufA, nd);
    kscatter<<<(unsigned)cdiv(escan, 256), 256, 0, stream>>>(src, dst, wn, h, bufA,
                                                             DINL, DINP, 1.0f, escan);
    kgemm_wmma<DINP, DOUT, LDC><<<gg, 128, 0, stream>>>(bufA, packB + hopstride, out,
                                                        total_tiles);

    const float* Tprev = bufA;   // T_{k-1}
    const float* Tprev2 = h;     // T_{k-2}
    for (int k = 2; k < K; ++k) {
        float* cur = (k & 1) ? bufA : bufB;     // k=2 -> bufB, k=3 -> bufA, ...
        // cur = -T_{k-2}  (for k>=3 this is an in-place negate)
        knegcopy<<<(unsigned)cdiv(nd, 256), 256, 0, stream>>>(cur, Tprev2, nd);
        // cur += 2 * L_hat @ T_{k-1}
        kscatter<<<(unsigned)cdiv(escan, 256), 256, 0, stream>>>(src, dst, wn, Tprev,
                                                                 cur, DINL, DINP, 2.0f, escan);
        kgemm_wmma<DINP, DOUT, LDC><<<gg, 128, 0, stream>>>(cur, packB + (size_t)k * hopstride,
                                                            out, total_tiles);
        Tprev2 = Tprev;
        Tprev = cur;
    }
}

// ---------------------------------------------------------------------------
// Entry point
// ---------------------------------------------------------------------------
extern "C" void kernel_launch(void* const* d_in, const int* in_sizes, int n_in,
                              void* d_out, int out_size, void* d_ws, size_t ws_size,
                              hipStream_t stream) {
    (void)in_sizes; (void)n_in; (void)out_size; (void)ws_size;

    const float* x    = (const float*)d_in[0];
    const int*   ei   = (const int*)d_in[1];
    const float* W1   = (const float*)d_in[2];
    const float* b1   = (const float*)d_in[3];
    const float* W2   = (const float*)d_in[4];
    const float* b2   = (const float*)d_in[5];
    const float* W3   = (const float*)d_in[6];
    const float* b3   = (const float*)d_in[7];
    const float* W4   = (const float*)d_in[8];
    const float* b4   = (const float*)d_in[9];
    const float* bn1g = (const float*)d_in[10];
    const float* bn1b = (const float*)d_in[11];
    const float* bn1m = (const float*)d_in[12];
    const float* bn1v = (const float*)d_in[13];
    const float* bn2g = (const float*)d_in[14];
    const float* bn2b = (const float*)d_in[15];
    const float* bn2m = (const float*)d_in[16];
    const float* bn2v = (const float*)d_in[17];
    const float* bn3g = (const float*)d_in[18];
    const float* bn3b = (const float*)d_in[19];
    const float* bn3m = (const float*)d_in[20];
    const float* bn3v = (const float*)d_in[21];

    const int* src = ei;            // edge_index[0]
    const int* dst = ei + N_EDGES;  // edge_index[1]

    // --- carve workspace (256-byte aligned regions) ---
    char* p = (char*)d_ws;
    auto carve = [&](size_t bytes) -> void* {
        void* r = (void*)p;
        p += (bytes + 255) & ~(size_t)255;
        return r;
    };
    float* deg   = (float*)carve((size_t)N_NODES * 4);
    float* dinv  = (float*)carve((size_t)N_NODES * 4);
    float* wn    = (float*)carve((size_t)N_EDGES * 4);
    float* bufA  = (float*)carve((size_t)N_NODES * 128 * 4);
    float* bufB  = (float*)carve((size_t)N_NODES * 128 * 4);
    float* h1    = (float*)carve((size_t)N_NODES * 64 * 4);   // stride 64
    float* h2    = (float*)carve((size_t)N_NODES * 32 * 4);   // logical 18, stride 32
    float* h3    = (float*)carve((size_t)N_NODES * 32 * 4);   // logical 9,  stride 32
    _Float16* packB = (_Float16*)carve((size_t)128 * 1024);   // max 128 KB (layer 1)

    // --- degree normalization ---
    kzero<<<(unsigned)cdiv(N_NODES, 256), 256, 0, stream>>>(deg, N_NODES);
    kdeg<<<(unsigned)cdiv(N_EDGES, 256), 256, 0, stream>>>(dst, deg, N_EDGES);
    kdinv<<<(unsigned)cdiv(N_NODES, 256), 256, 0, stream>>>(deg, dinv, N_NODES);
    kwnorm<<<(unsigned)cdiv(N_EDGES, 256), 256, 0, stream>>>(src, dst, dinv, wn, N_EDGES);

    // --- layer 1: 128 -> 64, K=8, BN+ReLU ---
    cheb_layer<128, 128, 64, 64>(x, W1, b1, 8, h1, src, dst, wn, bufA, bufB, packB, stream);
    kbnrelu<<<(unsigned)cdiv((long)N_NODES * 64, 256), 256, 0, stream>>>(
        h1, bn1g, bn1b, bn1m, bn1v, 64, 64, (long)N_NODES * 64);

    // --- layer 2: 64 -> 18 (stride 32), K=6, BN+ReLU ---
    cheb_layer<64, 64, 18, 32>(h1, W2, b2, 6, h2, src, dst, wn, bufA, bufB, packB, stream);
    kbnrelu<<<(unsigned)cdiv((long)N_NODES * 32, 256), 256, 0, stream>>>(
        h2, bn2g, bn2b, bn2m, bn2v, 18, 32, (long)N_NODES * 32);

    // --- layer 3: 18 (pad 32) -> 9 (stride 32), K=4, BN+ReLU ---
    cheb_layer<32, 18, 9, 32>(h2, W3, b3, 4, h3, src, dst, wn, bufA, bufB, packB, stream);
    kbnrelu<<<(unsigned)cdiv((long)N_NODES * 32, 256), 256, 0, stream>>>(
        h3, bn3g, bn3b, bn3m, bn3v, 9, 32, (long)N_NODES * 32);

    // --- layer 4: 9 (pad 32) -> 10 (exact stride, d_out), K=4 ---
    cheb_layer<32, 9, 10, 10>(h3, W4, b4, 4, (float*)d_out, src, dst, wn,
                              bufA, bufB, packB, stream);
}
